// Net_3831110828718
// MI455X (gfx1250) — compile-verified, matrix operand order checked by reading, not run on hardware
//
#include <hip/hip_runtime.h>
#include <hip/hip_bf16.h>
#include <math.h>

// ---------------------------------------------------------------------------
// SNN (eligibility-trace dead code removed):
//   per t: 3 LIF GEMMs (f16 WMMA, f32 acc) -> softmax error interneurons ->
//          2 feedback GEMMs (t>=soft_error_step)
// Single persistent workgroup (512 thr = 16 wave32 waves) loops over T.
// Layer1 GEMM uses 2x2 register blocking (A/B fragment reuse, 2 loads/wmma).
// Small cross-layer state lives in LDS (ds ops); big layer-1 state in global.
// ---------------------------------------------------------------------------

#define BATCH   64
#define INDIM   2312
#define TSTEPS  20
#define H1DIM   500
#define H2DIM   100
#define OUTDIM  10

#define KP1     2336   // INDIM padded to mult of 32
#define NP1     512    // H1 padded (also K of layer2)
#define KP2     512
#define NP2     128    // H2 padded (also K of layer3)
#define KP3     128
#define NP3     16     // OUT padded to 16
#define KPFB    32     // OUT padded to 32 (K of feedback GEMMs)
#define KT1     (KP1/32)
#define KT2     (KP2/32)
#define KT3     (KP3/32)

#define VDECAY  0.6f
#define VTH     0.3f

#define NTHREADS 512

typedef __attribute__((ext_vector_type(8)))  _Float16 v8h_t;
typedef __attribute__((ext_vector_type(16))) _Float16 v16h_t;
typedef __attribute__((ext_vector_type(8)))  float    v8f_t;

// ------------------------- pre-pass conversion kernels ----------------------

__global__ void cvt_pad_f16(const float* __restrict__ src, _Float16* __restrict__ dst,
                            int rows, int cols, int Rp, int Kp) {
    int idx = blockIdx.x * blockDim.x + threadIdx.x;
    int n = Rp * Kp;
    if (idx >= n) return;
    int r = idx / Kp, k = idx - r * Kp;
    float v = (r < rows && k < cols) ? src[r * cols + k] : 0.f;
    dst[idx] = (_Float16)v;
}

// spike_data (B, IN, T) f32 -> Xh (T, 64, KP1) f16, zero padded in K.
__global__ void cvt_x_f16(const float* __restrict__ x, _Float16* __restrict__ dst) {
    int idx = blockIdx.x * blockDim.x + threadIdx.x;
    const int n = TSTEPS * BATCH * KP1;
    if (idx >= n) return;
    int t = idx / (BATCH * KP1);
    int rem = idx - t * (BATCH * KP1);
    int m = rem / KP1, k = rem - m * KP1;
    float v = (k < INDIM) ? x[((size_t)m * INDIM + k) * TSTEPS + t] : 0.f;
    dst[idx] = (_Float16)v;
}

__global__ void zero_f32(float* __restrict__ p, int n) {
    int idx = blockIdx.x * blockDim.x + threadIdx.x;
    if (idx < n) p[idx] = 0.f;
}

// ------------------------- WMMA fragment helpers ----------------------------
// 16-bit fragment for lane L (ISA 7.12.2): row = tile*16 + (L%16);
// halves 0..7  = K : kbase + 8*(L/16) + [0..7]
// halves 8..15 = K : kbase + 8*(L/16) + 16 + [0..7]
// => two 16B chunks from one row of a row-major (rows x Kp) f16 matrix.

__device__ __forceinline__ v16h_t frag_at(const _Float16* p) {
    v8h_t lo = *(const v8h_t*)(p);
    v8h_t hi = *(const v8h_t*)(p + 16);
    v16h_t f;
#pragma unroll
    for (int i = 0; i < 8; ++i) { f[i] = lo[i]; f[i + 8] = hi[i]; }
    return f;
}

__device__ __forceinline__ const _Float16* row_ptr(const _Float16* base,
                                                   int tile, int Kp, int lane) {
    return base + (size_t)(tile * 16 + (lane & 15)) * Kp + ((lane >> 4) << 3);
}

// ------------------------- LIF layer: GEMM + neuron update ------------------
// act : (64 x Kp) f16 row-major      wgt : (Np x Kp) f16 row-major (B[k][n]=W[n][k])
// volt: (64 x Np) f32                spk : (64 x Np) f16 (prev spike in, new out)
// MB x NB register blocking of 16x16 output tiles (fragment reuse).
template <int MB, int NB>
__device__ __forceinline__ void lif_layer(const _Float16* act,
                                          const _Float16* wgt,
                                          int Kp, int ktiles, int ntiles, int H,
                                          const float* __restrict__ bias,
                                          float* volt,
                                          _Float16* spk,
                                          float* outAcc,
                                          int wave, int lane, int nwaves) {
    const int Np = ntiles * 16;
    const int mblocks = 4 / MB;                  // M = 64 -> 4 m-tiles
    const int nblocks = ntiles / NB;
    for (int blk = wave; blk < mblocks * nblocks; blk += nwaves) {
        const int mb = blk / nblocks;
        const int nb = blk - mb * nblocks;
        const _Float16* arow[MB];
        const _Float16* brow[NB];
#pragma unroll
        for (int i = 0; i < MB; ++i) arow[i] = row_ptr(act, mb * MB + i, Kp, lane);
#pragma unroll
        for (int j = 0; j < NB; ++j) brow[j] = row_ptr(wgt, nb * NB + j, Kp, lane);
        v8f_t acc[MB][NB];
#pragma unroll
        for (int i = 0; i < MB; ++i)
#pragma unroll
            for (int j = 0; j < NB; ++j) acc[i][j] = (v8f_t){};
        for (int kt = 0; kt < ktiles; ++kt) {
            v16h_t a[MB], b[NB];
#pragma unroll
            for (int i = 0; i < MB; ++i) a[i] = frag_at(arow[i] + kt * 32);
#pragma unroll
            for (int j = 0; j < NB; ++j) {
                b[j] = frag_at(brow[j] + kt * 32);
                __builtin_prefetch(brow[j] + kt * 32 + 32, 0, 1);  // next K-tile
            }
#pragma unroll
            for (int i = 0; i < MB; ++i)
#pragma unroll
                for (int j = 0; j < NB; ++j)
                    acc[i][j] = __builtin_amdgcn_wmma_f32_16x16x32_f16(
                        false, a[i], false, b[j], (short)0, acc[i][j], false, false);
        }
#pragma unroll
        for (int i = 0; i < MB; ++i) {
            const int mBase = (mb * MB + i) * 16 + ((lane >> 4) << 3);
#pragma unroll
            for (int j = 0; j < NB; ++j) {
                const int n = (nb * NB + j) * 16 + (lane & 15);
                const float bn = (n < H) ? bias[n] : 0.f;
#pragma unroll
                for (int r = 0; r < 8; ++r) {
                    const int m = mBase + r;
                    float*    vp = volt + (size_t)m * Np + n;
                    _Float16* sp = spk  + (size_t)m * Np + n;
                    const float sprev = (float)(*sp);
                    float v = VDECAY * (*vp) * (1.f - sprev) + acc[i][j][r] + bn;
                    float s = (v > VTH) ? 1.f : 0.f;
                    *vp = v;
                    *sp = (_Float16)s;
                    if (outAcc) outAcc[m * NP3 + n] += s;     // output layer only
                }
            }
        }
    }
}

// feedback GEMM: err += diff (64xKPFB) @ fb (Np x KPFB, row n = fb[n][:])
__device__ __forceinline__ void fb_layer(const _Float16* act,
                                         const _Float16* wgt,
                                         int ntiles, float* __restrict__ accOut,
                                         int wave, int lane, int nwaves) {
    const int Np = ntiles * 16;
    const int nTiles = 4 * ntiles;
    for (int tile = wave; tile < nTiles; tile += nwaves) {
        const int mt = tile / ntiles;
        const int nt = tile - mt * ntiles;
        v16h_t a = frag_at(row_ptr(act, mt, KPFB, lane));
        v16h_t b = frag_at(row_ptr(wgt, nt, KPFB, lane));
        v8f_t acc = {};
        acc = __builtin_amdgcn_wmma_f32_16x16x32_f16(
            false, a, false, b, (short)0, acc, false, false);
        const int n = nt * 16 + (lane & 15);
        const int mBase = mt * 16 + ((lane >> 4) << 3);
#pragma unroll
        for (int r = 0; r < 8; ++r)
            accOut[(size_t)(mBase + r) * Np + n] += acc[r];
    }
}

// ------------------------- persistent main kernel ---------------------------

__global__ __launch_bounds__(NTHREADS) void snn_main(
    const _Float16* __restrict__ Xh,
    const _Float16* __restrict__ W1h, const _Float16* __restrict__ W2h,
    const _Float16* __restrict__ Woh,
    const _Float16* __restrict__ fb1h, const _Float16* __restrict__ fb2h,
    const float* __restrict__ b1, const float* __restrict__ b2,
    const float* __restrict__ bo,
    const float* __restrict__ onehot, const int* __restrict__ softp,
    float* v1, float* v2,
    _Float16* s1,
    float* e1acc, float* e2acc,
    float* out) {
    // ---- persistent small state in LDS (lives across the whole time loop)
    __shared__ _Float16 s2sh[BATCH * NP2];     // layer-2 spikes (act of layer 3)
    __shared__ _Float16 sosh[BATCH * NP3];     // layer-3 prev spikes
    __shared__ float    vosh[BATCH * NP3];     // layer-3 volt
    __shared__ float    outSh[BATCH * NP3];    // output accumulator
    __shared__ float    epvSh[BATCH * NP3];
    __shared__ float    envSh[BATCH * NP3];
    __shared__ float    epsSh[BATCH * NP3];
    __shared__ float    ensSh[BATCH * NP3];
    __shared__ float    edvSh[BATCH * NP3];
    __shared__ _Float16 diffSh[BATCH * KPFB];  // error spikes (act of fb GEMMs)

    const int tid = threadIdx.x;
    const int lane = tid & 31;
    const int wave = tid >> 5;
    const int nwaves = blockDim.x >> 5;
    const int soft = *softp;

    for (int i = tid; i < BATCH * NP2; i += blockDim.x) s2sh[i] = (_Float16)0.f;
    for (int i = tid; i < BATCH * KPFB; i += blockDim.x) diffSh[i] = (_Float16)0.f;
    for (int i = tid; i < BATCH * NP3; i += blockDim.x) {
        sosh[i] = (_Float16)0.f;
        vosh[i] = 0.f; outSh[i] = 0.f;
        epvSh[i] = 0.f; envSh[i] = 0.f; epsSh[i] = 0.f; ensSh[i] = 0.f; edvSh[i] = 0.f;
    }
    __syncthreads();

    for (int t = 0; t < TSTEPS; ++t) {
        // layer 1: 2x2 blocked (32 blocks -> 2 per wave), K = 2336
        lif_layer<2, 2>(Xh + (size_t)t * BATCH * KP1, W1h, KP1, KT1, NP1 / 16, H1DIM,
                        b1, v1, s1, nullptr, wave, lane, nwaves);
        __syncthreads();
        // layer 2: 32 tiles -> 2 per wave
        lif_layer<1, 1>(s1, W2h, KP2, KT2, NP2 / 16, H2DIM,
                        b2, v2, s2sh, nullptr, wave, lane, nwaves);
        __syncthreads();
        // layer 3: acts from LDS, state in LDS
        lif_layer<1, 1>(s2sh, Woh, KP3, KT3, NP3 / 16, OUTDIM,
                        bo, vosh, sosh, outSh, wave, lane, nwaves);
        __syncthreads();

        if (t >= soft) {
            if (tid < BATCH) {
                const int m = tid;
                const float* orow = outSh + m * NP3;
                float mx = orow[0];
                for (int j = 1; j < OUTDIM; ++j) mx = fmaxf(mx, orow[j]);
                float ex[OUTDIM]; float sum = 0.f;
                for (int j = 0; j < OUTDIM; ++j) { ex[j] = expf(orow[j] - mx); sum += ex[j]; }
                const float inv = 1.f / sum;
                for (int j = 0; j < OUTDIM; ++j) {
                    const int q = m * NP3 + j;
                    float err = ex[j] * inv - onehot[m * OUTDIM + j];
                    float ep = fmaxf(err, 0.f), en = fmaxf(-err, 0.f);
                    float pvn = epvSh[q] - epsSh[q] + ep;
                    float ps = (pvn > 0.f) ? 1.f : 0.f;
                    float nvn = envSh[q] - ensSh[q] + en;
                    float ns = (nvn > 0.f) ? 1.f : 0.f;
                    float d = ps - ns;
                    epvSh[q] = pvn; envSh[q] = nvn; epsSh[q] = ps; ensSh[q] = ns;
                    edvSh[q] += d;
                    diffSh[m * KPFB + j] = (_Float16)d;   // cols 10..31 stay zero
                }
            }
            __syncthreads();
            fb_layer(diffSh, fb2h, NP2 / 16, e2acc, wave, lane, nwaves);
            fb_layer(diffSh, fb1h, NP1 / 16, e1acc, wave, lane, nwaves);
        }
        __syncthreads();
    }

    // ----- emit outputs: [predict(64)][output(640)][err1(32000)][err2(6400)][edv(640)]
    if (tid < BATCH) {
        const int m = tid;
        float best = outSh[m * NP3]; int bi = 0;
        for (int j = 1; j < OUTDIM; ++j) {
            float v = outSh[m * NP3 + j];
            if (v > best) { best = v; bi = j; }
        }
        out[m] = (float)bi;
        for (int j = 0; j < OUTDIM; ++j) {
            out[64 + m * OUTDIM + j]    = outSh[m * NP3 + j];
            out[39104 + m * OUTDIM + j] = edvSh[m * NP3 + j];
        }
    }
    for (int i = tid; i < BATCH * H1DIM; i += blockDim.x)
        out[704 + i] = e1acc[(size_t)(i / H1DIM) * NP1 + (i % H1DIM)];
    for (int i = tid; i < BATCH * H2DIM; i += blockDim.x)
        out[32704 + i] = e2acc[(size_t)(i / H2DIM) * NP2 + (i % H2DIM)];
}

// ------------------------------- host side ----------------------------------

extern "C" void kernel_launch(void* const* d_in, const int* in_sizes, int n_in,
                              void* d_out, int out_size, void* d_ws, size_t ws_size,
                              hipStream_t stream) {
    const float* spike_data = (const float*)d_in[0];   // (64, 2312, 20)
    const float* onehot     = (const float*)d_in[1];   // (64, 10)
    const float* W1  = (const float*)d_in[2];
    const float* b1  = (const float*)d_in[3];
    const float* W2  = (const float*)d_in[4];
    const float* b2  = (const float*)d_in[5];
    const float* Wo  = (const float*)d_in[6];
    const float* bo  = (const float*)d_in[7];
    const float* fb1 = (const float*)d_in[8];          // (500, 10)
    const float* fb2 = (const float*)d_in[9];          // (100, 10)
    const int*  soft = (const int*)d_in[10];
    float* out = (float*)d_out;

    char* ws = (char*)d_ws;
    size_t off = 0;
    auto alloc = [&](size_t bytes) -> void* {
        void* p = ws + off;
        off = (off + bytes + 255) & ~(size_t)255;
        return p;
    };

    _Float16* W1h  = (_Float16*)alloc((size_t)NP1 * KP1 * 2);
    _Float16* W2h  = (_Float16*)alloc((size_t)NP2 * KP2 * 2);
    _Float16* Woh  = (_Float16*)alloc((size_t)NP3 * KP3 * 2);
    _Float16* fb1h = (_Float16*)alloc((size_t)NP1 * KPFB * 2);
    _Float16* fb2h = (_Float16*)alloc((size_t)NP2 * KPFB * 2);
    _Float16* Xh   = (_Float16*)alloc((size_t)TSTEPS * BATCH * KP1 * 2);

    size_t stateStart = off;
    float*    v1    = (float*)   alloc((size_t)BATCH * NP1 * 4);
    float*    v2    = (float*)   alloc((size_t)BATCH * NP2 * 4);
    _Float16* s1    = (_Float16*)alloc((size_t)BATCH * NP1 * 2);
    float*    e1acc = (float*)   alloc((size_t)BATCH * NP1 * 4);
    float*    e2acc = (float*)   alloc((size_t)BATCH * NP2 * 4);
    size_t stateBytes = off - stateStart;

    const int TPB = 256;
    auto blocks = [](size_t n, int t) { return (unsigned)((n + t - 1) / t); };

    cvt_pad_f16<<<blocks((size_t)NP1 * KP1, TPB), TPB, 0, stream>>>(W1,  W1h,  H1DIM, INDIM, NP1, KP1);
    cvt_pad_f16<<<blocks((size_t)NP2 * KP2, TPB), TPB, 0, stream>>>(W2,  W2h,  H2DIM, H1DIM, NP2, KP2);
    cvt_pad_f16<<<blocks((size_t)NP3 * KP3, TPB), TPB, 0, stream>>>(Wo,  Woh,  OUTDIM, H2DIM, NP3, KP3);
    cvt_pad_f16<<<blocks((size_t)NP1 * KPFB, TPB), TPB, 0, stream>>>(fb1, fb1h, H1DIM, OUTDIM, NP1, KPFB);
    cvt_pad_f16<<<blocks((size_t)NP2 * KPFB, TPB), TPB, 0, stream>>>(fb2, fb2h, H2DIM, OUTDIM, NP2, KPFB);
    cvt_x_f16<<<blocks((size_t)TSTEPS * BATCH * KP1, TPB), TPB, 0, stream>>>(spike_data, Xh);

    zero_f32<<<blocks(stateBytes / 4, TPB), TPB, 0, stream>>>(
        (float*)(ws + stateStart), (int)(stateBytes / 4));

    snn_main<<<1, NTHREADS, 0, stream>>>(
        Xh, W1h, W2h, Woh, fb1h, fb2h,
        b1, b2, bo, onehot, soft,
        v1, v2, s1, e1acc, e2acc, out);
}
